// SparseResNet3D_72962904425000
// MI455X (gfx1250) — compile-verified, matrix operand order checked by reading, not run on hardware
//
#include <hip/hip_runtime.h>
#include <stdint.h>

// ---------------------------------------------------------------------------
// CDNA5 (gfx1250) masked 3D ResNet forward.
// bf16 WMMA (v_wmma_f32_16x16x32_bf16) for all 64-channel contractions;
// f32 master activations for BN/residual; bf16 operand copies for GEMMs;
// implicit-GEMM conv3 with TDM (tensor_load_to_lds) double-buffered LDS
// staging, synchronized with s_wait_tensorcnt.
// ---------------------------------------------------------------------------

typedef __bf16    v16bf __attribute__((ext_vector_type(16)));
typedef __bf16    v8bf  __attribute__((ext_vector_type(8)));
typedef float     v8f   __attribute__((ext_vector_type(8)));
typedef uint32_t  v4u   __attribute__((ext_vector_type(4)));
typedef uint32_t  v8u   __attribute__((ext_vector_type(8)));

__device__ __forceinline__ int kmap(int i, int h) {
    // A/B 16-bit fragment K mapping: lanes 0-15 (h=0): K={0..7,16..23},
    // lanes 16-31 (h=1): K={8..15,24..31}
    return (i < 8) ? (h * 8 + i) : (16 + h * 8 + (i - 8));
}

__device__ __forceinline__ v16bf join16(v8bf lo, v8bf hi) {
    return __builtin_shufflevector(lo, hi, 0, 1, 2, 3, 4, 5, 6, 7,
                                   8, 9, 10, 11, 12, 13, 14, 15);
}

__device__ __forceinline__ v8f wmma_bf16(v16bf a, v16bf b, v8f c) {
    return __builtin_amdgcn_wmma_f32_16x16x32_bf16(
        /*neg_a=*/false, a, /*neg_b=*/false, b,
        /*c_mod=*/(short)0, c, /*reuse_a=*/false, /*reuse_b=*/false);
}

// ---------------------------------------------------------------------------
// TDM: 2D tile load (tile1 rows x tile0 bf16 elements, row stride stride0
// elements) from global to LDS. Descriptor per CDNA5 ISA ch.8 (D# groups).
// Huge tensor dims disable far-edge OOB zeroing (boundaries fixed up by the
// caller). Issued by one wave; tracked with TENSORcnt.
// ---------------------------------------------------------------------------
__device__ __forceinline__ void tdm_load_2d(uint32_t lds_off, uint64_t gaddr,
                                            uint32_t tile0, uint32_t tile1,
                                            uint64_t stride0) {
    v4u g0;
    v8u g1;
    v4u g2 = {0, 0, 0, 0};
    v4u g3 = {0, 0, 0, 0};
    g0[0] = 1u;                                   // count=1 (valid), user mode
    g0[1] = lds_off;                              // lds_addr (bytes)
    g0[2] = (uint32_t)gaddr;                      // global_addr[31:0]
    g0[3] = (uint32_t)((gaddr >> 32) & 0x01FFFFFFu) | (2u << 30); // [120:64],type=2
    const uint32_t td0 = 0x40000000u;             // tensor_dim0 (huge)
    const uint32_t td1 = 0x40000000u;             // tensor_dim1 (huge)
    g1[0] = 1u << 16;                             // data_size=1 (2 bytes/elem)
    g1[1] = (td0 & 0xFFFFu) << 16;                // tensor_dim0[15:0]
    g1[2] = (td0 >> 16) | ((td1 & 0xFFFFu) << 16);// td0[31:16] | td1[15:0]
    g1[3] = (td1 >> 16) | (tile0 << 16);          // td1[31:16] | tile_dim0
    g1[4] = tile1 & 0xFFFFu;                      // tile_dim1 (tile_dim2=0)
    g1[5] = (uint32_t)stride0;                    // tensor_dim0_stride[31:0]
    g1[6] = (uint32_t)((stride0 >> 32) & 0xFFFFu);// stride0[47:32] (stride1=0)
    g1[7] = 0u;
    asm volatile("tensor_load_to_lds %0, %1, %2, %3"
                 :: "s"(g0), "s"(g1), "s"(g2), "s"(g3)
                 : "memory");
}

// ---------------------------------------------------------------------------
// Utility kernels
// ---------------------------------------------------------------------------
__global__ void k_zero(float* p, int n) {
    for (int i = blockIdx.x * blockDim.x + threadIdx.x; i < n;
         i += gridDim.x * blockDim.x)
        p[i] = 0.0f;
}

// maskf = (mask > 0.5); also writes maskf into d_out tail and accumulates n.
__global__ void k_maskf(const float* __restrict__ mask, float* __restrict__ maskf,
                        float* __restrict__ out_mask, float* __restrict__ nAccum,
                        int N) {
    __shared__ float sm[256];
    float acc = 0.0f;
    for (int v = blockIdx.x * blockDim.x + threadIdx.x; v < N;
         v += gridDim.x * blockDim.x) {
        float m = (mask[v] > 0.5f) ? 1.0f : 0.0f;
        maskf[v] = m;
        out_mask[v] = m;
        acc += m;
    }
    sm[threadIdx.x] = acc;
    __syncthreads();
    for (int s = 128; s > 0; s >>= 1) {
        if (threadIdx.x < s) sm[threadIdx.x] += sm[threadIdx.x + s];
        __syncthreads();
    }
    if (threadIdx.x == 0) atomicAdd(nAccum, sm[0]);
}

// ---------------------------------------------------------------------------
// Weight fragment conversion (f32 -> bf16, pre-swizzled to WMMA B layout).
// conv1-type W shape (co, 64): frag[((ks*ntN+nt)*32+l)*16+i] = W[o][c]
// ---------------------------------------------------------------------------
__global__ void k_wfrag1(const float* __restrict__ W, __bf16* __restrict__ frag,
                         int ntN) {
    int total = 2 * ntN * 512;
    for (int idx = blockIdx.x * blockDim.x + threadIdx.x; idx < total;
         idx += gridDim.x * blockDim.x) {
        int i = idx & 15;
        int l = (idx >> 4) & 31;
        int t = idx >> 9;           // t = ks*ntN + nt
        int nt = t % ntN;
        int ks = t / ntN;
        int o = nt * 16 + (l & 15);
        int c = ks * 32 + kmap(i, l >> 4);
        frag[idx] = (__bf16)W[o * 64 + c];
    }
}

// conv3 W shape (o=64, c=64, kd, kh, kw): tap = kd*9+kh*3+kw
// frag[tap*4096 + ks*2048 + nt*512 + l*16 + i] = W[o][c][tap]
__global__ void k_wfrag3(const float* __restrict__ W, __bf16* __restrict__ frag) {
    int total = 27 * 4096;
    for (int idx = blockIdx.x * blockDim.x + threadIdx.x; idx < total;
         idx += gridDim.x * blockDim.x) {
        int i = idx & 15;
        int l = (idx >> 4) & 31;
        int nt = (idx >> 9) & 3;
        int ks = (idx >> 11) & 1;
        int tap = idx >> 12;
        int o = nt * 16 + (l & 15);
        int c = ks * 32 + kmap(i, l >> 4);
        frag[idx] = (__bf16)W[((size_t)o * 64 + c) * 27 + tap];
    }
}

// ---------------------------------------------------------------------------
// Squeeze: 1x1 conv 4->64, masked input & output. Tiny K, plain VALU.
// ---------------------------------------------------------------------------
__global__ void k_squeeze(const float* __restrict__ feat,
                          const float* __restrict__ maskf,
                          const float* __restrict__ W,   // (64,4)
                          const float* __restrict__ bias,
                          float* __restrict__ out, int N, int vox) {
    for (int v = blockIdx.x * blockDim.x + threadIdx.x; v < N;
         v += gridDim.x * blockDim.x) {
        float mk = maskf[v];
        int b = v / vox;
        int vv = v - b * vox;
        const float* f = feat + (size_t)b * 4 * vox + vv;
        float x0 = f[0] * mk, x1 = f[vox] * mk;
        float x2 = f[2 * (size_t)vox] * mk, x3 = f[3 * (size_t)vox] * mk;
        float* o = out + (size_t)v * 64;
#pragma unroll
        for (int oc = 0; oc < 64; ++oc) {
            float y = W[oc * 4 + 0] * x0 + W[oc * 4 + 1] * x1 +
                      W[oc * 4 + 2] * x2 + W[oc * 4 + 3] * x3 + bias[oc];
            o[oc] = y * mk;
        }
    }
}

// ---------------------------------------------------------------------------
// 1x1 conv 64->64 via WMMA. One wave = 16 voxels x 64 out-chan (4 acc tiles).
// ---------------------------------------------------------------------------
__global__ void k_conv1_wmma(const __bf16* __restrict__ in,
                             const __bf16* __restrict__ wfrag,
                             const float* __restrict__ bias,
                             const float* __restrict__ maskf,
                             float* __restrict__ out) {
    int wave = threadIdx.x >> 5, lane = threadIdx.x & 31;
    int base = blockIdx.x * 128 + wave * 16;
    int m = lane & 15, half = lane >> 4;

    v8f acc[4] = {};
    const __bf16* arow = in + (size_t)(base + m) * 64;
#pragma unroll
    for (int ks = 0; ks < 2; ++ks) {
        v8bf a0 = *reinterpret_cast<const v8bf*>(arow + ks * 32 + half * 8);
        v8bf a1 = *reinterpret_cast<const v8bf*>(arow + ks * 32 + 16 + half * 8);
        v16bf af = join16(a0, a1);
#pragma unroll
        for (int nt = 0; nt < 4; ++nt) {
            v16bf bf = *reinterpret_cast<const v16bf*>(
                wfrag + (((ks * 4 + nt) * 32 + lane) << 4));
            acc[nt] = wmma_bf16(af, bf, acc[nt]);
        }
    }
    int mb = half * 8;
#pragma unroll
    for (int nt = 0; nt < 4; ++nt) {
        int ch = nt * 16 + m;
        float bv = bias[ch];
#pragma unroll
        for (int r = 0; r < 8; ++r) {
            int vox = base + mb + r;
            float val = (acc[nt][r] + bv) * maskf[vox];
            out[(size_t)vox * 64 + ch] = val;
        }
    }
}

// ---------------------------------------------------------------------------
// 3x3x3 SAME conv 64->64, implicit GEMM. One block per (b,d,h) output line
// (48 voxels = 3 waves x 16). TDM stages 3 h-lines x 50 w x 64ch bf16 per kd
// into double-buffered LDS; wave0 issues the next tile while all waves run
// 72 WMMAs on the current one. Halo columns / OOB rows are zero-fixed after
// s_wait_tensorcnt (tile rows start at w=-1; input buffer has guard margins).
// ---------------------------------------------------------------------------
__global__ void k_conv3_wmma(const __bf16* __restrict__ in,
                             const __bf16* __restrict__ wfrag,
                             const float* __restrict__ bias,
                             const float* __restrict__ maskf,
                             float* __restrict__ out) {
    __shared__ __align__(16) __bf16 lds[2][3 * 50 * 64];  // 2 x 19.2 KB
    const int Dd = 48, Hh = 48, Ww = 48;
    int line = blockIdx.x;  // 0 .. B*D*H-1
    int h = line % Hh;
    int t = line / Hh;
    int d = t % Dd;
    int b = t / Dd;
    int lane = threadIdx.x & 31, wave = threadIdx.x >> 5;
    int wbase = wave * 16;
    int m = lane & 15, half = lane >> 4;
    size_t lineBase = (((size_t)b * Dd + d) * Hh + h) * Ww;

    // --- TDM issue helper: 3 rows x 3200 elems, row stride = one h-line ---
    auto issue = [&](int jd, int buf) {
        int dd = d + jd - 1;
        long long rowElem =
            ((((long long)b * Dd + dd) * Hh + (h - 1)) * (long long)Ww - 1) * 64;
        uint64_t gaddr = (uint64_t)(uintptr_t)in + (long long)rowElem * 2;
        uint32_t lds_off = (uint32_t)(uintptr_t)&lds[buf][0];
        tdm_load_2d(lds_off, gaddr, /*tile0=*/3200, /*tile1=*/3,
                    /*stride0=*/(uint64_t)(Hh * 0 + 48 * 64));
    };

    // --- fixup: zero halo columns (always x-OOB) and invalid rows ---
    auto fixup = [&](int jd, int buf) {
        int dd = d + jd - 1;
        bool dok = (dd >= 0 && dd < Dd);
        v8bf z = {};
        for (int c = threadIdx.x; c < 48; c += blockDim.x) {  // 2 cols x 3 rows
            int jh = c >> 4;
            int rem = c & 15;
            int col = (rem >> 3) ? 49 : 0;
            int chb = (rem & 7) * 8;
            *reinterpret_cast<v8bf*>(&lds[buf][(jh * 50 + col) * 64 + chb]) = z;
        }
#pragma unroll
        for (int jh = 0; jh < 3; ++jh) {
            int hh = h + jh - 1;
            if (!dok || hh < 0 || hh >= Hh) {
                for (int c = threadIdx.x; c < 400; c += blockDim.x)
                    *reinterpret_cast<v8bf*>(&lds[buf][jh * 3200 + c * 8]) = z;
            }
        }
    };

    v8f acc[4] = {};
    if (threadIdx.x < 32) issue(0, 0);

    for (int jd = 0; jd < 3; ++jd) {  // kd tap, input depth d+jd-1
        int buf = jd & 1;
        if (threadIdx.x < 32) {
            if (jd < 2) {
                issue(jd + 1, buf ^ 1);
                __builtin_amdgcn_s_wait_tensorcnt(1);  // tile jd complete
            } else {
                __builtin_amdgcn_s_wait_tensorcnt(0);
            }
        }
        __syncthreads();
        fixup(jd, buf);
        __syncthreads();

        const __bf16* wf = wfrag + (size_t)jd * 9 * 4096;
        const __bf16* lbuf = &lds[buf][0];
#pragma unroll
        for (int jh = 0; jh < 3; ++jh) {
#pragma unroll
            for (int kw = 0; kw < 3; ++kw) {
                int widx = wbase + m + kw;  // 0..49
                const __bf16* ar = lbuf + (jh * 50 + widx) * 64;
                const __bf16* wt = wf + ((jh * 3 + kw) * 4096);
#pragma unroll
                for (int ks = 0; ks < 2; ++ks) {
                    v8bf a0 = *reinterpret_cast<const v8bf*>(ar + ks * 32 + half * 8);
                    v8bf a1 = *reinterpret_cast<const v8bf*>(ar + ks * 32 + 16 + half * 8);
                    v16bf af = join16(a0, a1);
#pragma unroll
                    for (int nt = 0; nt < 4; ++nt) {
                        v16bf bf = *reinterpret_cast<const v16bf*>(
                            wt + (((ks * 4 + nt) * 32 + lane) << 4));
                        acc[nt] = wmma_bf16(af, bf, acc[nt]);
                    }
                }
            }
        }
        __syncthreads();  // buf is TDM-overwritten in the next iteration
    }

    int mb = half * 8;
#pragma unroll
    for (int nt = 0; nt < 4; ++nt) {
        int ch = nt * 16 + m;
        float bv = bias[ch];
#pragma unroll
        for (int r = 0; r < 8; ++r) {
            size_t v = lineBase + wbase + mb + r;
            float val = (acc[nt][r] + bv) * maskf[v];
            out[v * 64 + ch] = val;
        }
    }
}

// ---------------------------------------------------------------------------
// Final 1x1 conv 64->32, WMMA, writes NCDHW to d_out (no mask, per reference).
// ---------------------------------------------------------------------------
__global__ void k_final_wmma(const __bf16* __restrict__ in,
                             const __bf16* __restrict__ wfrag,
                             const float* __restrict__ bias,
                             float* __restrict__ out, int vox) {
    int wave = threadIdx.x >> 5, lane = threadIdx.x & 31;
    int base = blockIdx.x * 128 + wave * 16;
    int m = lane & 15, half = lane >> 4;

    v8f acc[2] = {};
    const __bf16* arow = in + (size_t)(base + m) * 64;
#pragma unroll
    for (int ks = 0; ks < 2; ++ks) {
        v8bf a0 = *reinterpret_cast<const v8bf*>(arow + ks * 32 + half * 8);
        v8bf a1 = *reinterpret_cast<const v8bf*>(arow + ks * 32 + 16 + half * 8);
        v16bf af = join16(a0, a1);
#pragma unroll
        for (int nt = 0; nt < 2; ++nt) {
            v16bf bf = *reinterpret_cast<const v16bf*>(
                wfrag + (((ks * 2 + nt) * 32 + lane) << 4));
            acc[nt] = wmma_bf16(af, bf, acc[nt]);
        }
    }
    int mb = half * 8;
#pragma unroll
    for (int nt = 0; nt < 2; ++nt) {
        int ch = nt * 16 + m;
        float bv = bias[ch];
#pragma unroll
        for (int r = 0; r < 8; ++r) {
            int v = base + mb + r;
            int bb = v / vox;
            int vv = v - bb * vox;
            out[((size_t)bb * 32 + ch) * vox + vv] = acc[nt][r] + bv;
        }
    }
}

// ---------------------------------------------------------------------------
// BatchNorm (masked): stats -> finalize -> apply (fused mask/lrelu/residual)
// stats layout: [0..63]=sum, [64..127]=sumsq, [128..191]=scale,
//               [192..255]=shift, [256]=n
// ---------------------------------------------------------------------------
__global__ void k_bn_stats(const float* __restrict__ x, float* __restrict__ stats,
                           int N) {
    __shared__ float ss[256], sq[256];
    int ch = threadIdx.x & 63, grp = threadIdx.x >> 6;
    float s = 0.0f, q = 0.0f;
    for (int v = blockIdx.x * 4 + grp; v < N; v += gridDim.x * 4) {
        float val = x[(size_t)v * 64 + ch];
        s += val;
        q += val * val;
    }
    ss[threadIdx.x] = s;
    sq[threadIdx.x] = q;
    __syncthreads();
    if (grp == 0) {
        s = ss[ch] + ss[64 + ch] + ss[128 + ch] + ss[192 + ch];
        q = sq[ch] + sq[64 + ch] + sq[128 + ch] + sq[192 + ch];
        atomicAdd(&stats[ch], s);
        atomicAdd(&stats[64 + ch], q);
    }
}

__global__ void k_bn_finalize(float* __restrict__ stats,
                              const float* __restrict__ g,
                              const float* __restrict__ beta) {
    int ch = threadIdx.x;
    if (ch >= 64) return;
    float n = stats[256];
    float mean = stats[ch] / n;
    float var = stats[64 + ch] / n - mean * mean;
    float inv = rsqrtf(var + 1e-5f);
    float sc = g[ch] * inv;
    stats[128 + ch] = sc;
    stats[192 + ch] = beta[ch] - mean * sc;
}

__global__ void k_bn_apply(const float* __restrict__ x,
                           const float* __restrict__ stats,
                           const float* __restrict__ maskf,
                           const float* __restrict__ residual,
                           float* __restrict__ out_f32,
                           __bf16* __restrict__ out_bf, int total,
                           int do_lrelu) {
    for (int idx = blockIdx.x * blockDim.x + threadIdx.x; idx < total;
         idx += gridDim.x * blockDim.x) {
        int ch = idx & 63;
        int v = idx >> 6;
        float y = (x[idx] * stats[128 + ch] + stats[192 + ch]) * maskf[v];
        if (residual) y += residual[idx];
        if (do_lrelu) y = (y > 0.0f) ? y : 0.01f * y;
        if (out_f32) out_f32[idx] = y;
        out_bf[idx] = (__bf16)y;
    }
}

// ---------------------------------------------------------------------------
// Host orchestration
// ---------------------------------------------------------------------------
extern "C" void kernel_launch(void* const* d_in, const int* in_sizes, int n_in,
                              void* d_out, int out_size, void* d_ws,
                              size_t ws_size, hipStream_t stream) {
    (void)n_in; (void)out_size; (void)ws_size;
    const float* feat = (const float*)d_in[0];
    const float* mask = (const float*)d_in[1];
    const float* sqW = (const float*)d_in[2];
    const float* sqb = (const float*)d_in[3];
    const float* sqg = (const float*)d_in[4];
    const float* sqbeta = (const float*)d_in[5];
    const float *c1W[4], *c1b[4], *g1[4], *be1[4];
    const float *c2W[4], *c2b[4], *g2[4], *be2[4];
    const float *c3W[4], *c3b[4], *g3[4], *be3[4];
    for (int i = 0; i < 4; ++i) {
        int base = 6 + i * 12;
        c1W[i] = (const float*)d_in[base + 0];
        c1b[i] = (const float*)d_in[base + 1];
        g1[i] = (const float*)d_in[base + 2];
        be1[i] = (const float*)d_in[base + 3];
        c2W[i] = (const float*)d_in[base + 4];
        c2b[i] = (const float*)d_in[base + 5];
        g2[i] = (const float*)d_in[base + 6];
        be2[i] = (const float*)d_in[base + 7];
        c3W[i] = (const float*)d_in[base + 8];
        c3b[i] = (const float*)d_in[base + 9];
        g3[i] = (const float*)d_in[base + 10];
        be3[i] = (const float*)d_in[base + 11];
    }
    const float* fW = (const float*)d_in[54];
    const float* fb = (const float*)d_in[55];

    const int N = in_sizes[1];   // B*D*H*W = 221184
    const int vox = N / 2;       // 110592
    const int lines = N / 48;    // B*D*H

    char* ws = (char*)d_ws;
    size_t off = 0;
    auto carve = [&](size_t bytes) {
        size_t o = off;
        off = (off + bytes + 255) & ~(size_t)255;
        return o;
    };
    float* maskf = (float*)(ws + carve((size_t)N * 4));
    float* stats = (float*)(ws + carve(4096));
    float* Xf = (float*)(ws + carve((size_t)N * 64 * 4));
    float* Hf = (float*)(ws + carve((size_t)N * 64 * 4));
    __bf16* XbA = (__bf16*)(ws + carve((size_t)N * 64 * 2));
    carve(512 * 1024);  // guard: TDM tile rows may read up to ~300KB outside
    __bf16* XbB = (__bf16*)(ws + carve((size_t)N * 64 * 2));
    carve(512 * 1024);  // guard
    __bf16 *fr_c1[4], *fr_c2[4], *fr_c3[4];
    for (int i = 0; i < 4; ++i) {
        fr_c1[i] = (__bf16*)(ws + carve(8192));
        fr_c2[i] = (__bf16*)(ws + carve(221184));
        fr_c3[i] = (__bf16*)(ws + carve(8192));
    }
    __bf16* fr_f = (__bf16*)(ws + carve(4096));

    float* outp = (float*)d_out;
    float* outMask = outp + (size_t)32 * N;  // B*32*vox = 32*N

    // ---- init & mask ----
    k_zero<<<2, 256, 0, stream>>>(stats, 1024);
    k_maskf<<<432, 256, 0, stream>>>(mask, maskf, outMask, &stats[256], N);

    // ---- weight fragment conversion (bf16, B-layout) ----
    for (int i = 0; i < 4; ++i) {
        k_wfrag1<<<16, 256, 0, stream>>>(c1W[i], fr_c1[i], 4);
        k_wfrag3<<<432, 256, 0, stream>>>(c2W[i], fr_c2[i]);
        k_wfrag1<<<16, 256, 0, stream>>>(c3W[i], fr_c3[i], 4);
    }
    k_wfrag1<<<8, 256, 0, stream>>>(fW, fr_f, 2);

    auto bn = [&](const float* xbuf, const float* g, const float* beta,
                  const float* residual, float* outf, __bf16* outbf,
                  int lrelu) {
        k_zero<<<1, 256, 0, stream>>>(stats, 128);
        k_bn_stats<<<1024, 256, 0, stream>>>(xbuf, stats, N);
        k_bn_finalize<<<1, 64, 0, stream>>>(stats, g, beta);
        k_bn_apply<<<4096, 256, 0, stream>>>(xbuf, stats, maskf, residual,
                                             outf, outbf, N * 64, lrelu);
    };

    // ---- squeeze: conv1(4->64) + BN + lrelu ----
    k_squeeze<<<864, 256, 0, stream>>>(feat, maskf, sqW, sqb, Xf, N, vox);
    bn(Xf, sqg, sqbeta, nullptr, Xf, XbA, 1);

    // ---- 4 residual blocks ----
    for (int i = 0; i < 4; ++i) {
        k_conv1_wmma<<<N / 128, 256, 0, stream>>>(XbA, fr_c1[i], c1b[i], maskf, Hf);
        bn(Hf, g1[i], be1[i], nullptr, nullptr, XbB, 1);
        k_conv3_wmma<<<lines, 96, 0, stream>>>(XbB, fr_c2[i], c2b[i], maskf, Hf);
        bn(Hf, g2[i], be2[i], nullptr, nullptr, XbB, 1);
        k_conv1_wmma<<<N / 128, 256, 0, stream>>>(XbB, fr_c3[i], c3b[i], maskf, Hf);
        bn(Hf, g3[i], be3[i], Xf, Xf, XbA, 1);  // bn3 + residual + lrelu
    }

    // ---- final 1x1 conv 64->32, NCDHW output ----
    k_final_wmma<<<N / 128, 256, 0, stream>>>(XbA, fr_f, fb, outp, vox);
}
